// FBPinn_17300128268889
// MI455X (gfx1250) — matrix operand order repaired and checked by reading, not compiled, not run on hardware
//
#include <hip/hip_runtime.h>
#include <hip/hip_bf16.h>

// ---------------------------------------------------------------------------
// FBPinn 1D forward on MI455X (gfx1250, wave32, WMMA).
// Compute-bound (27 GFLOP vs ~1.3 MB traffic): hidden 64x64 layers go through
// v_wmma_f32_16x16x32_f16; the runtime is dominated by ~307M tanh/sigmoid
// evaluations, so those must lower to native trans ops (v_tanh_f32 if the
// builtin exists, else v_exp + v_rcp), never IEEE division expansions.
// ---------------------------------------------------------------------------

typedef __attribute__((ext_vector_type(16))) _Float16 v16h;
typedef __attribute__((ext_vector_type(8)))  _Float16 v8h;
typedef __attribute__((ext_vector_type(8)))  float    v8f;

#define NW      16
#define NEURONS 64
#define PTS_PER_BLOCK 128     // 8 waves * 16 points
#define STRA 72               // activation row stride in halves (16B aligned, spreads banks)
#define STRK 72               // transposed-weight row stride in halves

__device__ __forceinline__ float fast_rcp(float x) {
    return __builtin_amdgcn_rcpf(x);          // single v_rcp_f32
}

__device__ __forceinline__ float tanh_fast(float x) {
#if __has_builtin(__builtin_amdgcn_tanhf)
    return __builtin_amdgcn_tanhf(x);         // native v_tanh_f32 on gfx1250
#else
    // tanh(x) = 1 - 2/(exp(2x)+1); saturates correctly at +-inf
    float e = __expf(2.0f * x);
    return 1.0f - 2.0f * fast_rcp(e + 1.0f);
#endif
}
__device__ __forceinline__ float sigmoid_fast(float z) {
    return fast_rcp(1.0f + __expf(-z));
}

// Build a 16-half WMMA fragment from two aligned 16B LDS chunks.
__device__ __forceinline__ v16h ld_frag(const _Float16* p0, const _Float16* p1) {
    v8h lo = *(const v8h*)p0;
    v8h hi = *(const v8h*)p1;
    v16h r;
#pragma unroll
    for (int i = 0; i < 8; ++i) { r[i] = lo[i]; r[i + 8] = hi[i]; }
    return r;
}

__global__ __launch_bounds__(256) void fbpinn_fwd(
    const float* __restrict__ x,      // [N]
    const float* __restrict__ W_in,   // [NW,64]
    const float* __restrict__ b_in,   // [NW,64]
    const float* __restrict__ W_h,    // [2,NW,64,64] (i=in, o=out)
    const float* __restrict__ b_h,    // [2,NW,64]
    const float* __restrict__ W_out,  // [NW,64]
    const float* __restrict__ b_out,  // [NW]
    float* __restrict__ out,          // [N]
    int N)
{
    __shared__ __align__(16) _Float16 act_s[8 * 16 * STRA];   // per-wave 16x64 activation tiles
    __shared__ __align__(16) _Float16 wt_s[2 * 64 * STRK];    // W_h transposed: [layer][n][k] f16
    __shared__ float xs[PTS_PER_BLOCK];
    __shared__ float Win_s[NEURONS], bin_s[NEURONS], Wout_s[NEURONS];
    __shared__ float bh_s[2 * NEURONS];

    const int tid  = threadIdx.x;
    const int wave = tid >> 5;
    const int lane = tid & 31;
    const int col  = lane & 15;            // row m for A / column n for B,C,D
    const int grp  = (lane >> 4) & 1;      // lane group (K-half selector)

    const int block_base = blockIdx.x * PTS_PER_BLOCK;

    // Stage x for this block (clamped OOB -> garbage compute, guarded store).
    if (tid < PTS_PER_BLOCK) {
        int g = block_base + tid;
        if (g > N - 1) g = N - 1;
        xs[tid] = x[g];
    }

    _Float16* actw = act_s + wave * 16 * STRA;  // this wave's 16-point tile

    float pred = 0.0f;                     // lanes 0..15 of each wave own one point

    const float width = 6.25f;             // (100-0)/16

    for (int w = 0; w < NW; ++w) {
        // ---- window geometry (uniform, scalar) ----
        const float lo_w  = (w == 0)      ? 0.0f   : ((float)w - 0.125f) * width;
        const float hi_w  = (w == NW - 1) ? 100.0f : ((float)w + 1.125f) * width;
        const float mean  = 0.5f * (lo_w + hi_w);
        const float inv_s = 2.0f * fast_rcp(hi_w - lo_w);
        const float ovm0  = (float)w * width;
        const float ovm1  = (float)(w + 1) * width;
        const float bout  = b_out[w];

        // ---- cooperative weight staging for this window ----
        __syncthreads();   // previous iteration done with LDS weights
        for (int c = tid; c < 2 * NEURONS * NEURONS; c += 256) {
            int l   = c >> 12;
            int rem = c & 4095;
            int i   = rem >> 6;      // input index  (k)
            int o   = rem & 63;      // output index (n)  -> contiguous global reads
            wt_s[(l * 64 + o) * STRK + i] =
                (_Float16)W_h[(((l * NW) + w) * 64 + i) * 64 + o];
        }
        if (tid < NEURONS) {
            Win_s[tid]  = W_in[w * 64 + tid];
            bin_s[tid]  = b_in[w * 64 + tid];
            Wout_s[tid] = W_out[w * 64 + tid];
        }
        if (tid >= 128 && tid < 256) {
            int c = tid - 128;
            int l = c >> 6, o = c & 63;
            bh_s[c] = b_h[((l * NW) + w) * 64 + o];
        }
        __syncthreads();

        // ---- input layer: act[p][j] = tanh(xn[p]*W_in[j] + b_in[j]) ----
        for (int e = lane; e < 16 * NEURONS; e += 32) {
            int p = e >> 6, j = e & 63;
            float xv = xs[wave * 16 + p];
            float xn = (xv - mean) * inv_s;
            actw[p * STRA + j] = (_Float16)tanh_fast(fmaf(xn, Win_s[j], bin_s[j]));
        }

        // ---- hidden layers: [16x64] = tanh([16x64]x[64x64] + b) via WMMA ----
#pragma unroll
        for (int layer = 0; layer < 2; ++layer) {
            const _Float16* Wt = wt_s + layer * 64 * STRK;

            // A fragments: row m=col; halves 0..7 at K = kc + grp*8,
            //              halves 8..15 at K = kc + 16 + grp*8  (ISA 16-bit A layout)
            const _Float16* rowA = actw + col * STRA;
            v16h a0 = ld_frag(rowA + 0  + grp * 8, rowA + 16 + grp * 8);
            v16h a1 = ld_frag(rowA + 32 + grp * 8, rowA + 48 + grp * 8);

#pragma unroll
            for (int nt = 0; nt < 4; ++nt) {
                // B fragments: column n = nt*16+col; halves 0..7 at K = kc + grp*8, 8..15 at K = kc + 16 + grp*8
                const _Float16* rowB = Wt + (nt * 16 + col) * STRK;
                v16h b0 = ld_frag(rowB + grp * 8,      rowB + 16 + grp * 8);
                v16h b1 = ld_frag(rowB + 32 + grp * 8, rowB + 48 + grp * 8);

                v8f c = {};
                c = __builtin_amdgcn_wmma_f32_16x16x32_f16(
                        false, a0, false, b0, (short)0, c, false, false);
                c = __builtin_amdgcn_wmma_f32_16x16x32_f16(
                        false, a1, false, b1, (short)0, c, false, false);

                // epilogue: bias + tanh, write back per C layout
                float bias = bh_s[layer * 64 + nt * 16 + col];
#pragma unroll
                for (int r = 0; r < 8; ++r) {
                    float t = tanh_fast(c[r] + bias);
                    actw[(r + grp * 8) * STRA + nt * 16 + col] = (_Float16)t;
                }
            }
        }

        // ---- output layer + window weighting (lanes 0..15 own point p=lane) ----
        if (lane < 16) {
            const _Float16* row = actw + lane * STRA;
            float s = 0.0f;
#pragma unroll
            for (int ch = 0; ch < 8; ++ch) {          // 8 x ds_load_b128
                v8h v = *(const v8h*)(row + ch * 8);
#pragma unroll
                for (int i = 0; i < 8; ++i)
                    s = fmaf((float)v[i], Wout_s[ch * 8 + i], s);
            }
            float o  = s + bout;
            float xv = xs[wave * 16 + lane];
            float win = sigmoid_fast((xv - ovm0) * 2.0f) *
                        sigmoid_fast(-(xv - ovm1) * 2.0f);
            bool inside = (lo_w < xv) && (xv < hi_w);
            pred += inside ? (win * o) : 0.0f;
        }
    }

    if (lane < 16) {
        int g = block_base + wave * 16 + lane;
        if (g < N) out[g] = pred;
    }
}

extern "C" void kernel_launch(void* const* d_in, const int* in_sizes, int n_in,
                              void* d_out, int out_size, void* d_ws, size_t ws_size,
                              hipStream_t stream) {
    const float* x     = (const float*)d_in[0];
    const float* W_in  = (const float*)d_in[1];
    const float* b_in  = (const float*)d_in[2];
    const float* W_h   = (const float*)d_in[3];
    const float* b_h   = (const float*)d_in[4];
    const float* W_out = (const float*)d_in[5];
    const float* b_out = (const float*)d_in[6];
    float* out = (float*)d_out;

    int N = in_sizes[0];
    int blocks = (N + PTS_PER_BLOCK - 1) / PTS_PER_BLOCK;
    fbpinn_fwd<<<blocks, 256, 0, stream>>>(x, W_in, b_in, W_h, b_h, W_out, b_out,
                                           out, N);
    (void)d_ws; (void)ws_size; (void)out_size; (void)n_in;
}
